// Attention_6932077215914
// MI455X (gfx1250) — matrix operand order repaired and checked by reading, not compiled, not run on hardware
//
#include <hip/hip_runtime.h>

typedef unsigned short u16;
typedef __attribute__((ext_vector_type(16))) __bf16 v16bf;
typedef __attribute__((ext_vector_type(8)))  float  v8f;
typedef __attribute__((ext_vector_type(4)))  unsigned int v4u;

// ---- problem constants ----
static constexpr int B_      = 2;
static constexpr int SQ_     = 2048;
static constexpr int HID     = 2048;
static constexpr int HEADS_  = 16;
static constexpr int GROUPS_ = 4;
static constexpr int KVC_    = 128;
static constexpr float SCALE_F = 0.08838834764831845f; // 128^-0.5

__device__ __forceinline__ u16 f2bf(float f) {
  union { float f; unsigned int u; } v; v.f = f;
  unsigned int u = v.u;
  unsigned int rb = 0x7FFFu + ((u >> 16) & 1u);
  return (u16)((u + rb) >> 16);
}

union FragU { v16bf bf; v4u q[2]; };

__device__ __forceinline__ v16bf ld_frag(const u16* p0, const u16* p1) {
  FragU f;
  f.q[0] = *(const v4u*)p0;
  f.q[1] = *(const v4u*)p1;
  return f.bf;
}

// CDNA5 async global->LDS copy (ASYNCcnt-tracked), 16B per lane.
// lds  : LDS byte offset (low 32 bits of generic shared pointer)
// goff : per-lane byte offset from uniform base
__device__ __forceinline__ void async_b128(unsigned lds, unsigned goff, const void* base) {
  asm volatile("global_load_async_to_lds_b128 %0, %1, %2"
               :: "v"(lds), "v"(goff), "s"(base) : "memory");
}
__device__ __forceinline__ void wait_async0() {
  asm volatile("s_wait_asynccnt 0x0" ::: "memory");
}
__device__ __forceinline__ unsigned lds_off(const void* p) {
  return (unsigned)(uintptr_t)p;   // generic LDS addr: low 32 bits = LDS offset
}

// =====================================================================
// f32 -> bf16 elementwise convert (vectorized, one-shot preprocessing)
// =====================================================================
__global__ __launch_bounds__(256)
void cvt_bf16(const float* __restrict__ in, u16* __restrict__ out) {
  long i = ((long)blockIdx.x * 256 + threadIdx.x) * 4;
  float4 v = *(const float4*)(in + i);
  uint2 r;
  r.x = (unsigned)f2bf(v.x) | ((unsigned)f2bf(v.y) << 16);
  r.y = (unsigned)f2bf(v.z) | ((unsigned)f2bf(v.w) << 16);
  *(uint2*)(out + i) = r;
}

// =====================================================================
// W[K][N] f32 -> Wt[N][K] bf16 (tiled transpose, one-shot preprocessing)
// =====================================================================
__global__ __launch_bounds__(256)
void transpose_cvt(const float* __restrict__ w, u16* __restrict__ wT, int K, int N) {
  __shared__ float t[32][33];
  int n0 = blockIdx.x * 32, k0 = blockIdx.y * 32;
  int tx = threadIdx.x & 31, ty = threadIdx.x >> 5;  // 32 x 8
  for (int i = 0; i < 4; ++i)
    t[ty + i * 8][tx] = w[(long)(k0 + ty + i * 8) * N + n0 + tx];
  __syncthreads();
  for (int i = 0; i < 4; ++i)
    wT[(long)(n0 + ty + i * 8) * K + k0 + tx] = f2bf(t[tx][ty + i * 8]);
}

// =====================================================================
// GEMM: C[M,N] = A[M,K] * Bt[N,K]^T ; bf16 inputs, f32 accumulate.
// 256 threads = 8 waves, 128x128 tile, K-step 32, async double-buffered.
// EPI: 0 = f32 C0 ; 1 = bf16 C0 ; 2 = KV split (bf16 K->C0, bf16 V^T->C1)
// =====================================================================
template <int EPI>
__global__ __launch_bounds__(256)
void gemm_bf16(const u16* __restrict__ A, const u16* __restrict__ Bt,
               void* __restrict__ C0, void* __restrict__ C1,
               int M, int N, int K) {
  __shared__ u16 As[2][128][40];   // [buf][m][k], 80B rows
  __shared__ u16 Bs[2][128][40];   // [buf][n][k]

  const int tid  = threadIdx.x;
  const int wave = tid >> 5;
  const int lane = tid & 31;
  const int hi   = lane >> 4;
  const int l16  = lane & 15;
  const int wm = (wave & 3) * 32;
  const int wn = (wave >> 2) * 64;
  const long m0 = (long)blockIdx.y * 128;
  const long n0 = (long)blockIdx.x * 128;

  const unsigned abase = lds_off(&As[0][0][0]);
  const unsigned bbase = lds_off(&Bs[0][0][0]);

  v8f acc[2][4];
  for (int im = 0; im < 2; ++im)
    for (int in = 0; in < 4; ++in)
      for (int j = 0; j < 8; ++j) acc[im][in][j] = 0.f;

  auto stage = [&](int buf, int k0) {
#pragma unroll
    for (int i = 0; i < 2; ++i) {
      int c = tid + 256 * i;           // 0..511
      int r = c >> 2, q = c & 3;       // row, 16B chunk
      unsigned ga = (unsigned)(((m0 + r) * (long)K + k0) * 2) + q * 16;
      async_b128(abase + buf * 10240 + r * 80 + q * 16, ga, A);
      unsigned gb = (unsigned)(((n0 + r) * (long)K + k0) * 2) + q * 16;
      async_b128(bbase + buf * 10240 + r * 80 + q * 16, gb, Bt);
    }
  };

  stage(0, 0);
  wait_async0();
  __syncthreads();

  int buf = 0;
  for (int k0 = 0; k0 < K; k0 += 32) {
    if (k0 + 32 < K) stage(buf ^ 1, k0 + 32);

    v16bf af[2], bfv[4];
#pragma unroll
    for (int im = 0; im < 2; ++im) {
      const u16* p = &As[buf][wm + im * 16 + l16][hi * 8];
      af[im] = ld_frag(p, p + 16);
    }
#pragma unroll
    for (int in = 0; in < 4; ++in) {
      const u16* p = &Bs[buf][wn + in * 16 + l16][hi * 8];
      bfv[in] = ld_frag(p, p + 16);
    }
#pragma unroll
    for (int im = 0; im < 2; ++im)
#pragma unroll
      for (int in = 0; in < 4; ++in)
        acc[im][in] = __builtin_amdgcn_wmma_f32_16x16x32_bf16(
            false, af[im], false, bfv[in], (short)0, acc[im][in], false, false);

    wait_async0();
    __syncthreads();
    buf ^= 1;
  }

  for (int im = 0; im < 2; ++im)
    for (int in = 0; in < 4; ++in)
      for (int j = 0; j < 8; ++j) {
        long r = m0 + wm + im * 16 + j + hi * 8;
        long c = n0 + wn + in * 16 + l16;
        float v = acc[im][in][j];
        if (EPI == 0) {
          ((float*)C0)[r * N + c] = v;
        } else if (EPI == 1) {
          ((u16*)C0)[r * N + c] = f2bf(v);
        } else {
          if (c < 512) {               // K part: [m][512] row-major
            ((u16*)C0)[r * 512 + c] = f2bf(v);
          } else {                     // V part: transposed [b,g,d][s]
            int e = (int)c - 512;
            int g = e >> 7, d = e & 127;
            int bb = (int)(r >> 11), s = (int)(r & 2047);
            ((u16*)C1)[((long)(bb * GROUPS_ + g) * 128 + d) * SQ_ + s] = f2bf(v);
          }
        }
      }
}

// =====================================================================
// Flash attention (causal, GQA): block = (b, head, 64 q rows)
// 128 threads = 4 waves; d = 128; key tile = 64; all operands bf16.
// qb: [B*SQ][2048] bf16 ; kb: [B*SQ][512] bf16 ; vt: [B*G*128][SQ] bf16
// ctx out: [B*SQ][2048] bf16
// =====================================================================
__global__ __launch_bounds__(128)
void attn_flash(const u16* __restrict__ qb, const u16* __restrict__ kb,
                const u16* __restrict__ vt, u16* __restrict__ ctx) {
  __shared__ u16 Qs[64][136];    // [qrow][d], 272B rows
  __shared__ u16 Ks[64][136];    // [key][d]
  __shared__ u16 VsT[128][72];   // [d][key], 144B rows
  __shared__ u16 Ps[4][16][72];  // per-wave P tile

  const int tid  = threadIdx.x;
  const int wave = tid >> 5;
  const int lane = tid & 31;
  const int hi   = lane >> 4;
  const int l16  = lane & 15;

  const int q0 = blockIdx.x * 64;
  const int h  = blockIdx.y;
  const int b  = blockIdx.z;
  const int g  = h >> 2;

  const unsigned qso = lds_off(&Qs[0][0]);
  const unsigned kso = lds_off(&Ks[0][0]);
  const unsigned vso = lds_off(&VsT[0][0]);

  // stage Q tile (async): 64 rows x 256B
#pragma unroll
  for (int i = 0; i < 8; ++i) {
    int c = tid + 128 * i;
    int r = c >> 4, qc = c & 15;
    unsigned ga = (unsigned)((((long)(b * SQ_ + q0 + r)) * 2048 + h * 128) * 2) + qc * 16;
    async_b128(qso + r * 272 + qc * 16, ga, qb);
  }
  wait_async0();
  __syncthreads();

  v16bf qf[4];
#pragma unroll
  for (int c = 0; c < 4; ++c) {
    const u16* p = &Qs[wave * 16 + l16][c * 32 + hi * 8];
    qf[c] = ld_frag(p, p + 16);
  }

  v8f O[8];
  for (int dt = 0; dt < 8; ++dt)
    for (int j = 0; j < 8; ++j) O[dt][j] = 0.f;
  float mrow[8], lrow[8];
  for (int j = 0; j < 8; ++j) { mrow[j] = -1e30f; lrow[j] = 0.f; }

  const int nkt = q0 / 64 + 1;
  for (int kt = 0; kt < nkt; ++kt) {
    const int kbase = kt * 64;
    __syncthreads();   // all waves done reading previous K/V tiles
    // stage K tile (row-major) and V tile (pre-transposed) via async copies
#pragma unroll
    for (int i = 0; i < 8; ++i) {
      int c = tid + 128 * i;
      int r = c >> 4, qc = c & 15;
      unsigned ga = (unsigned)((((long)(b * SQ_ + kbase + r)) * 512 + g * 128) * 2) + qc * 16;
      async_b128(kso + r * 272 + qc * 16, ga, kb);
    }
#pragma unroll
    for (int i = 0; i < 8; ++i) {
      int c = tid + 128 * i;
      int r = c >> 3, qc = c & 7;
      unsigned ga = (unsigned)((((long)(b * GROUPS_ + g) * 128 + r) * SQ_ + kbase) * 2) + qc * 16;
      async_b128(vso + r * 144 + qc * 16, ga, vt);
    }
    wait_async0();
    __syncthreads();

    // S = Q * K^T  (16 x 64 per wave)
    v8f S[4];
#pragma unroll
    for (int in = 0; in < 4; ++in) {
      for (int j = 0; j < 8; ++j) S[in][j] = 0.f;
#pragma unroll
      for (int c = 0; c < 4; ++c) {
        const u16* p = &Ks[in * 16 + l16][c * 32 + hi * 8];
        v16bf kbf = ld_frag(p, p + 16);
        S[in] = __builtin_amdgcn_wmma_f32_16x16x32_bf16(
            false, qf[c], false, kbf, (short)0, S[in], false, false);
      }
    }

    // scale + causal mask
    const int rowg = q0 + wave * 16 + hi * 8;
    for (int in = 0; in < 4; ++in) {
      int col = kbase + in * 16 + l16;
      for (int j = 0; j < 8; ++j) {
        float s = S[in][j] * SCALE_F;
        S[in][j] = (col <= rowg + j) ? s : -1e30f;
      }
    }

    // online softmax
    for (int j = 0; j < 8; ++j) {
      float mx = fmaxf(fmaxf(S[0][j], S[1][j]), fmaxf(S[2][j], S[3][j]));
      for (int o = 8; o >= 1; o >>= 1) mx = fmaxf(mx, __shfl_xor(mx, o, 32));
      float mn = fmaxf(mrow[j], mx);
      float corr = __expf(mrow[j] - mn);
      mrow[j] = mn;
      lrow[j] *= corr;
      for (int dt = 0; dt < 8; ++dt) O[dt][j] *= corr;
      float rs = 0.f;
      for (int in = 0; in < 4; ++in) {
        float p = __expf(S[in][j] - mn);
        S[in][j] = p;
        rs += p;
      }
      for (int o = 8; o >= 1; o >>= 1) rs += __shfl_xor(rs, o, 32);
      lrow[j] += rs;
    }

    // C-layout -> A-layout via per-wave LDS bounce
    for (int in = 0; in < 4; ++in)
      for (int j = 0; j < 8; ++j)
        Ps[wave][j + hi * 8][in * 16 + l16] = f2bf(S[in][j]);
    asm volatile("s_wait_dscnt 0x0" ::: "memory");

    v16bf pf[2];
#pragma unroll
    for (int kc = 0; kc < 2; ++kc) {
      const u16* p = &Ps[wave][l16][kc * 32 + hi * 8];
      pf[kc] = ld_frag(p, p + 16);
    }

    // O += P * V
#pragma unroll
    for (int dt = 0; dt < 8; ++dt)
#pragma unroll
      for (int kc = 0; kc < 2; ++kc) {
        const u16* p = &VsT[dt * 16 + l16][kc * 32 + hi * 8];
        v16bf vb = ld_frag(p, p + 16);
        O[dt] = __builtin_amdgcn_wmma_f32_16x16x32_bf16(
            false, pf[kc], false, vb, (short)0, O[dt], false, false);
      }
  }

  // normalize, write bf16 ctx
  for (int dt = 0; dt < 8; ++dt)
    for (int j = 0; j < 8; ++j) {
      int qq = q0 + wave * 16 + j + hi * 8;
      float o = O[dt][j] / lrow[j];
      ctx[((long)(b * SQ_ + qq)) * 2048 + h * 128 + dt * 16 + l16] = f2bf(o);
    }
}

// =====================================================================
extern "C" void kernel_launch(void* const* d_in, const int* in_sizes, int n_in,
                              void* d_out, int out_size, void* d_ws, size_t ws_size,
                              hipStream_t stream) {
  (void)in_sizes; (void)n_in; (void)out_size; (void)ws_size;
  const float* x       = (const float*)d_in[0];
  const float* w_q     = (const float*)d_in[1];
  const float* w_kv    = (const float*)d_in[2];
  const float* w_dense = (const float*)d_in[3];
  float* out = (float*)d_out;

  const int M = B_ * SQ_;  // 4096
  u16* xb   = (u16*)d_ws;                       // [4096][2048]
  u16* wqT  = xb   + (size_t)M * HID;           // [2048][2048]
  u16* wkvT = wqT  + (size_t)HID * HID;         // [1024][2048]
  u16* wdT  = wkvT + (size_t)1024 * HID;        // [2048][2048]
  u16* qb   = wdT  + (size_t)HID * HID;         // [4096][2048]
  u16* kb   = qb   + (size_t)M * HID;           // [4096][512]
  u16* vt   = kb   + (size_t)M * 512;           // [1024][2048]
  u16* ctxb = vt   + (size_t)1024 * SQ_;        // [4096][2048]

  // ---- one-shot preprocessing: convert / transpose to bf16 ----
  cvt_bf16<<<(M * HID) / 1024, 256, 0, stream>>>(x, xb);
  transpose_cvt<<<dim3(HID / 32, HID / 32), 256, 0, stream>>>(w_q, wqT, HID, HID);
  transpose_cvt<<<dim3(1024 / 32, HID / 32), 256, 0, stream>>>(w_kv, wkvT, HID, 1024);
  transpose_cvt<<<dim3(HID / 32, HID / 32), 256, 0, stream>>>(w_dense, wdT, HID, HID);

  // ---- Q projection -> bf16 ----
  gemm_bf16<1><<<dim3(HID / 128, M / 128), 256, 0, stream>>>(
      xb, wqT, qb, nullptr, M, HID, HID);
  // ---- KV projection -> split bf16 K + transposed bf16 V ----
  gemm_bf16<2><<<dim3(1024 / 128, M / 128), 256, 0, stream>>>(
      xb, wkvT, kb, vt, M, 1024, HID);
  // ---- attention -> bf16 ctx ----
  attn_flash<<<dim3(SQ_ / 64, HEADS_, B_), 128, 0, stream>>>(qb, kb, vt, ctxb);
  // ---- output projection -> f32 out ----
  gemm_bf16<0><<<dim3(HID / 128, M / 128), 256, 0, stream>>>(
      ctxb, wdT, out, nullptr, M, HID, HID);
}
